// MLA_81106162418389
// MI455X (gfx1250) — compile-verified, matrix operand order checked by reading, not run on hardware
//
#include <hip/hip_runtime.h>
#include <hip/hip_bf16.h>

// ---------------------------------------------------------------------------
// MLA projection pipeline for MI455X (gfx1250, wave32, WMMA).
// GEMMs: v_wmma_f32_16x16x32_bf16; double-buffered LDS tiles staged by the
// Tensor Data Mover (tensor_load_to_lds + s_wait_tensorcnt) when available,
// else global_load_async_to_lds_b128 (ASYNCcnt). 4 waves x 64x64 per tile.
// ---------------------------------------------------------------------------

#define TRY_TDM 1
#if TRY_TDM && __has_builtin(__builtin_amdgcn_tensor_load_to_lds)
#define USE_TDM 1
#else
#define USE_TDM 0
#endif

typedef __bf16 bf16_t;
typedef __bf16 v16bf __attribute__((ext_vector_type(16)));
typedef float  v8f   __attribute__((ext_vector_type(8)));
typedef unsigned int u32x4 __attribute__((ext_vector_type(4)));
typedef int          i32x8 __attribute__((ext_vector_type(8)));
typedef int          i32x4 __attribute__((ext_vector_type(4)));

#define HID_DIM 2048
#define Q_LR    768
#define KV_LR   512
#define QK_NOPE 128
#define QK_ROPE 64
#define QK_HEAD 192
#define NHEAD   16
#define V_DIM   128
#define SEQ     4096
#define NROWS   8192            // B*S = 2*4096
#define OUT_D   512             // 192 + 192 + 128
#define LOG2_THETA 13.287712379549449f   // log2(10000)

// ------------------------------ bf16 convert -------------------------------
__global__ __launch_bounds__(256)
void cvt_bf16_kernel(const float* __restrict__ in, bf16_t* __restrict__ out, long n8)
{
    long i = (long)blockIdx.x * blockDim.x + threadIdx.x;
    long stride = (long)gridDim.x * blockDim.x;
    for (; i < n8; i += stride) {
        float4 f0 = reinterpret_cast<const float4*>(in)[i * 2];
        float4 f1 = reinterpret_cast<const float4*>(in)[i * 2 + 1];
        union { bf16_t h[8]; uint4 u; } p;
        p.h[0] = (bf16_t)f0.x; p.h[1] = (bf16_t)f0.y;
        p.h[2] = (bf16_t)f0.z; p.h[3] = (bf16_t)f0.w;
        p.h[4] = (bf16_t)f1.x; p.h[5] = (bf16_t)f1.y;
        p.h[6] = (bf16_t)f1.z; p.h[7] = (bf16_t)f1.w;
        reinterpret_cast<uint4*>(out)[i] = p.u;
    }
}

// ------------------------------ staging helpers ----------------------------
__device__ __forceinline__ unsigned lds_off(const void* p)
{
    // LDS byte offset = low 32 bits of the flat shared-aperture address
    return (unsigned)(unsigned long long)p;
}

__device__ __forceinline__ void async_copy_b128(const bf16_t* gsrc, bf16_t* lds_dst)
{
    unsigned loff = lds_off(lds_dst);
    unsigned long long ga = (unsigned long long)gsrc;
    asm volatile("global_load_async_to_lds_b128 %0, %1, off"
                 :: "v"(loff), "v"(ga) : "memory");
}

__device__ __forceinline__ void wait_async0()
{
#if __has_builtin(__builtin_amdgcn_s_wait_asynccnt)
    __builtin_amdgcn_s_wait_asynccnt(0);
#else
    asm volatile("s_wait_asynccnt 0" ::: "memory");
#endif
}

#if USE_TDM
// TDM: move one [rows<=128] x 32 bf16 tile (row stride K elements) into LDS
// laid out with 40-half row pitch (64B data + 16B pad -> pad_interval=16 DW,
// pad_amount=4 DW). Rows beyond rows_rem are zero-filled by TDM OOB handling.
__device__ __forceinline__ void tdm_load_tile(unsigned long long gaddr_bytes,
                                              unsigned lds_byte_off,
                                              unsigned rows_rem, unsigned K)
{
    u32x4 g0;
    g0[0] = 1u;                                       // count=1, user descriptor
    g0[1] = lds_byte_off;                             // lds_addr
    g0[2] = (unsigned)gaddr_bytes;                    // global_addr[31:0]
    g0[3] = (unsigned)(gaddr_bytes >> 32) | (2u << 30); // addr[56:32] | type=2
    i32x8 g1;
    g1[0] = (int)((1u << 16)      // data_size = 2 bytes
                | (1u << 20)      // pad_enable
                | (3u << 22)      // pad_interval: 16 DWORDs (64B) between pads
                | (3u << 25));    // pad_amount:   4 DWORDs (16B)
    g1[1] = (int)((K & 0xFFFFu) << 16);               // tensor_dim0[15:0]
    g1[2] = (int)((rows_rem & 0xFFFFu) << 16);        // dim0 hi=0 | tensor_dim1 lo
    g1[3] = (int)((rows_rem >> 16) | (32u << 16));    // dim1 hi | tile_dim0=32
    g1[4] = 128;                                      // tile_dim1=128, tile_dim2=0
    g1[5] = (int)K;                                   // tensor_dim0_stride lo32
    g1[6] = 0;                                        // stride hi | dim1_stride lo
    g1[7] = 0;
    i32x4 z4; z4[0] = 0; z4[1] = 0; z4[2] = 0; z4[3] = 0;
    i32x8 z8;
    #pragma unroll
    for (int i = 0; i < 8; ++i) z8[i] = 0;
    // clang-23 / amdgpu-toolchain 6-arg form:
    // (u32x4 g0, i32x8 g1, i32x4 g2, i32x4 g3, i32x8 pad, i32 cpol)
    __builtin_amdgcn_tensor_load_to_lds(g0, g1, z4, z4, z8, 0);
}

__device__ __forceinline__ void wait_tensor0()
{
    __builtin_amdgcn_s_wait_tensorcnt(0);
}
#endif

// ------------------------------ WMMA GEMM ----------------------------------
// C[M,N] = A[M,K] @ W[N,K]^T + bias  (A,W bf16 row-major K-contiguous)
// MODE 0: plain fp32 store; MODE 1: q-proj scatter; MODE 2: kv-proj scatter
#define BM 128
#define BN 128
#define BKK 32
#define LDT 40                  // LDS row stride in bf16 (32 + 8 pad)
#define GEMM_THREADS 128        // 4 waves; wave tile 64x64

union FragU { uint4 u[2]; v16bf v; };

#if !USE_TDM
// per-thread async staging: 512 x 16B segments, 4 per thread
__device__ __forceinline__ void stage_tile(const bf16_t* __restrict__ base,
                                           int row_clamp, int K, int k0, int row0,
                                           bf16_t* __restrict__ lbuf, int tid)
{
    #pragma unroll
    for (int p = 0; p < 4; ++p) {
        const int q   = p * GEMM_THREADS + tid;
        const int row = q >> 2;
        const int seg = (q & 3) * 8;
        int gr = row0 + row;
        gr = (gr < row_clamp) ? gr : (row_clamp - 1);
        async_copy_b128(base + (size_t)gr * K + k0 + seg, lbuf + row * LDT + seg);
    }
}
#endif

__device__ __forceinline__ void compute_step(const bf16_t* __restrict__ sAb,
                                             const bf16_t* __restrict__ sWb,
                                             int wm0, int wn0, int lo, int hi,
                                             v8f (&acc)[4][4])
{
    // ISA 16-bit A-fragment layout: lane<16 row=lo K 0..7 / 16..23,
    // lane>=16 row=lo K 8..15 / 24..31  -> two 16B ds reads per fragment.
    v16bf af[4], wf[4];
    #pragma unroll
    for (int i = 0; i < 4; ++i) {
        const bf16_t* p = sAb + (wm0 + i * 16 + lo) * LDT + hi * 8;
        FragU f;
        f.u[0] = *reinterpret_cast<const uint4*>(p);
        f.u[1] = *reinterpret_cast<const uint4*>(p + 16);
        af[i] = f.v;
    }
    #pragma unroll
    for (int j = 0; j < 4; ++j) {
        const bf16_t* p = sWb + (wn0 + j * 16 + lo) * LDT + hi * 8;
        FragU f;
        f.u[0] = *reinterpret_cast<const uint4*>(p);
        f.u[1] = *reinterpret_cast<const uint4*>(p + 16);
        wf[j] = f.v;
    }
    #pragma unroll
    for (int i = 0; i < 4; ++i)
        #pragma unroll
        for (int j = 0; j < 4; ++j)
            acc[i][j] = __builtin_amdgcn_wmma_f32_16x16x32_bf16(
                false, af[i], false, wf[j], (short)0, acc[i][j], false, false);
}

template<int MODE>
__global__ __launch_bounds__(GEMM_THREADS)
void gemm_wmma(const bf16_t* __restrict__ A, const bf16_t* __restrict__ W,
               const float* __restrict__ bias,
               float* __restrict__ Cplain, float* __restrict__ out,
               float* __restrict__ qrope,
               int M, int N, int K)
{
    __shared__ __align__(16) bf16_t sA[2][BM * LDT];
    __shared__ __align__(16) bf16_t sW[2][BN * LDT];

    const int tid  = threadIdx.x;
    const int lane = tid & 31;
    const int wid  = tid >> 5;                // 0..3
    const int wm0  = (wid >> 1) * 64;         // 2x2 wave grid, 64x64 per wave
    const int wn0  = (wid & 1) * 64;
    const int lo   = lane & 15;
    const int hi   = lane >> 4;
    const int bm   = blockIdx.y * BM;
    const int bn   = blockIdx.x * BN;

    v8f acc[4][4];
    #pragma unroll
    for (int i = 0; i < 4; ++i)
        #pragma unroll
        for (int j = 0; j < 4; ++j)
            #pragma unroll
            for (int r = 0; r < 8; ++r) acc[i][j][r] = 0.0f;

    const int nk = K / BKK;

    // prologue: stage tile 0 into buffer 0
#if USE_TDM
    if (wid == 0) {
        tdm_load_tile((unsigned long long)(A + (size_t)bm * K),
                      lds_off(sA[0]), (unsigned)(M - bm), (unsigned)K);
        tdm_load_tile((unsigned long long)(W + (size_t)bn * K),
                      lds_off(sW[0]), (unsigned)(N - bn), (unsigned)K);
    }
    wait_tensor0();   // no-op for waves with no outstanding TDM ops
#else
    stage_tile(A, M, K, 0, bm, sA[0], tid);
    stage_tile(W, N, K, 0, bn, sW[0], tid);
    wait_async0();
#endif
    __syncthreads();

    for (int kt = 0; kt < nk; ++kt) {
        const int cur = kt & 1;
        // overlap: stage next tile into the other buffer while computing
#if USE_TDM
        if (wid == 0 && kt + 1 < nk) {
            const int k1 = (kt + 1) * BKK;
            tdm_load_tile((unsigned long long)(A + (size_t)bm * K + k1),
                          lds_off(sA[cur ^ 1]), (unsigned)(M - bm), (unsigned)K);
            tdm_load_tile((unsigned long long)(W + (size_t)bn * K + k1),
                          lds_off(sW[cur ^ 1]), (unsigned)(N - bn), (unsigned)K);
        }
#else
        if (kt + 1 < nk) {
            stage_tile(A, M, K, (kt + 1) * BKK, bm, sA[cur ^ 1], tid);
            stage_tile(W, N, K, (kt + 1) * BKK, bn, sW[cur ^ 1], tid);
        }
#endif
        compute_step(sA[cur], sW[cur], wm0, wn0, lo, hi, acc);
#if USE_TDM
        wait_tensor0();   // unconditional: TENSORcnt==0 already for waves 1..3
#else
        wait_async0();
#endif
        __syncthreads();
    }

    // ---- epilogue: bias + mode-specific scatter ----
    #pragma unroll
    for (int i = 0; i < 4; ++i) {
        #pragma unroll
        for (int j = 0; j < 4; ++j) {
            #pragma unroll
            for (int r = 0; r < 8; ++r) {
                const int gm = bm + wm0 + i * 16 + hi * 8 + r;   // row (b*S+s)
                const int gn = bn + wn0 + j * 16 + lo;           // output column
                if (gn < N) {
                    float val = acc[i][j][r] + bias[gn];
                    if constexpr (MODE == 0) {
                        Cplain[(size_t)gm * N + gn] = val;
                    } else if constexpr (MODE == 1) {
                        const int h = gn / QK_HEAD, d = gn % QK_HEAD;
                        const int b = gm >> 12, s = gm & (SEQ - 1);
                        if (d < QK_NOPE) {
                            out[((size_t)(b * NHEAD + h) * SEQ + s) * OUT_D + d] = val;
                        } else {
                            qrope[(size_t)gm * (NHEAD * QK_ROPE) + h * QK_ROPE + (d - QK_NOPE)] = val;
                        }
                    } else {
                        const int h = gn >> 8, d = gn & 255;
                        const int b = gm >> 12, s = gm & (SEQ - 1);
                        const size_t ob = ((size_t)(b * NHEAD + h) * SEQ + s) * OUT_D;
                        if (d < QK_NOPE) out[ob + QK_HEAD + d] = val;                 // k_nope
                        else             out[ob + 2 * QK_HEAD + (d - QK_NOPE)] = val; // v
                    }
                }
            }
        }
    }
}

// ------------------------------ LayerNorm -> bf16 --------------------------
template<int NC, int STRIDE>
__global__ __launch_bounds__(256)
void ln_kernel(const float* __restrict__ x, const float* __restrict__ g,
               const float* __restrict__ b, bf16_t* __restrict__ y)
{
    __shared__ float red[256];
    const int row = blockIdx.x, tid = threadIdx.x;
    constexpr int PT = NC / 256;
    float v[PT];
    float s = 0.0f;
    #pragma unroll
    for (int i = 0; i < PT; ++i) {
        v[i] = x[(size_t)row * STRIDE + tid + i * 256];
        s += v[i];
    }
    red[tid] = s; __syncthreads();
    for (int o = 128; o > 0; o >>= 1) { if (tid < o) red[tid] += red[tid + o]; __syncthreads(); }
    const float mu = red[0] / (float)NC;
    __syncthreads();
    float q = 0.0f;
    #pragma unroll
    for (int i = 0; i < PT; ++i) { float d = v[i] - mu; q += d * d; }
    red[tid] = q; __syncthreads();
    for (int o = 128; o > 0; o >>= 1) { if (tid < o) red[tid] += red[tid + o]; __syncthreads(); }
    const float inv = rsqrtf(red[0] / (float)NC + 1e-5f);
    #pragma unroll
    for (int i = 0; i < PT; ++i) {
        const int c = tid + i * 256;
        y[(size_t)row * NC + c] = (bf16_t)((v[i] - mu) * inv * g[c] + b[c]);
    }
}

// ------------------------------ k RoPE (broadcast to 16 heads) -------------
__global__ __launch_bounds__(64)
void krot_kernel(const float* __restrict__ kvc, float* __restrict__ out)
{
    const int row = blockIdx.x;          // 0..8191
    const int d   = threadIdx.x;         // 0..63
    const int b = row >> 12, s = row & (SEQ - 1);
    const int j = d & 31;
    const float f  = (float)s * exp2f(-(float)j * (LOG2_THETA / 32.0f));
    const float c  = __cosf(f), sn = __sinf(f);
    const float* kr = kvc + (size_t)row * (KV_LR + QK_ROPE) + KV_LR;
    const float x  = kr[d];
    const float xp = (d < 32) ? -kr[d + 32] : kr[d - 32];
    const float val = x * c + xp * sn;
    size_t base = ((size_t)(b * NHEAD) * SEQ + s) * OUT_D + 2 * QK_HEAD - QK_ROPE + d; // col 320+d
    #pragma unroll
    for (int h = 0; h < NHEAD; ++h)
        out[base + (size_t)h * SEQ * OUT_D] = val;
}

// ------------------------------ q RoPE -------------------------------------
__global__ __launch_bounds__(256)
void qrope_kernel(const float* __restrict__ qr, float* __restrict__ out)
{
    const int row = blockIdx.x;
    const int b = row >> 12, s = row & (SEQ - 1);
    #pragma unroll
    for (int it = 0; it < 4; ++it) {
        const int idx = threadIdx.x + it * 256;    // 0..1023
        const int h = idx >> 6, d = idx & 63, j = d & 31;
        const float f  = (float)s * exp2f(-(float)j * (LOG2_THETA / 32.0f));
        const float c  = __cosf(f), sn = __sinf(f);
        const float* xr = qr + (size_t)row * (NHEAD * QK_ROPE) + h * QK_ROPE;
        const float x  = xr[d];
        const float xp = (d < 32) ? -xr[d + 32] : xr[d - 32];
        out[((size_t)(b * NHEAD + h) * SEQ + s) * OUT_D + QK_NOPE + d] = x * c + xp * sn;
    }
}

// ------------------------------ launch -------------------------------------
extern "C" void kernel_launch(void* const* d_in, const int* in_sizes, int n_in,
                              void* d_out, int out_size, void* d_ws, size_t ws_size,
                              hipStream_t stream)
{
    const float* hidden  = (const float*)d_in[0];
    const float* w_qa    = (const float*)d_in[1];
    const float* b_qa    = (const float*)d_in[2];
    const float* g_qa_ln = (const float*)d_in[3];
    const float* b_qa_ln = (const float*)d_in[4];
    const float* w_qb    = (const float*)d_in[5];
    const float* b_qb    = (const float*)d_in[6];
    const float* w_kva   = (const float*)d_in[7];
    const float* b_kva   = (const float*)d_in[8];
    const float* g_kva_ln= (const float*)d_in[9];
    const float* b_kva_ln= (const float*)d_in[10];
    const float* w_kvb   = (const float*)d_in[11];
    const float* b_kvb   = (const float*)d_in[12];
    float* out = (float*)d_out;

    // workspace carve-up (bytes)
    char* ws = (char*)d_ws;
    size_t off = 0;
    auto take = [&](size_t bytes) { char* p = ws + off; off += (bytes + 255) & ~(size_t)255; return p; };
    bf16_t* hid_bf  = (bf16_t*)take((size_t)NROWS * HID_DIM * 2);
    bf16_t* wqa_bf  = (bf16_t*)take((size_t)Q_LR * HID_DIM * 2);
    bf16_t* wkva_bf = (bf16_t*)take((size_t)(KV_LR + QK_ROPE) * HID_DIM * 2);
    bf16_t* wqb_bf  = (bf16_t*)take((size_t)(NHEAD * QK_HEAD) * Q_LR * 2);
    bf16_t* wkvb_bf = (bf16_t*)take((size_t)(NHEAD * (QK_NOPE + V_DIM)) * KV_LR * 2);
    float*  qc_f32  = (float*)take((size_t)NROWS * Q_LR * 4);
    float*  kvc_f32 = (float*)take((size_t)NROWS * (KV_LR + QK_ROPE) * 4);
    bf16_t* qcln_bf = (bf16_t*)take((size_t)NROWS * Q_LR * 2);
    bf16_t* kvlr_bf = (bf16_t*)take((size_t)NROWS * KV_LR * 2);
    float*  qrope_f = (float*)take((size_t)NROWS * NHEAD * QK_ROPE * 4);
    (void)ws_size; (void)in_sizes; (void)n_in; (void)out_size;

    // 1) fp32 -> bf16 conversions
    cvt_bf16_kernel<<<2048, 256, 0, stream>>>(hidden, hid_bf, (long)NROWS * HID_DIM / 8);
    cvt_bf16_kernel<<<512, 256, 0, stream>>>(w_qa,  wqa_bf,  (long)Q_LR * HID_DIM / 8);
    cvt_bf16_kernel<<<512, 256, 0, stream>>>(w_kva, wkva_bf, (long)(KV_LR + QK_ROPE) * HID_DIM / 8);
    cvt_bf16_kernel<<<512, 256, 0, stream>>>(w_qb,  wqb_bf,  (long)(NHEAD * QK_HEAD) * Q_LR / 8);
    cvt_bf16_kernel<<<512, 256, 0, stream>>>(w_kvb, wkvb_bf, (long)(NHEAD * (QK_NOPE + V_DIM)) * KV_LR / 8);

    // 2) down-projections: q_c and kv_c (WMMA, plain fp32 store)
    gemm_wmma<0><<<dim3(Q_LR / BN, NROWS / BM), GEMM_THREADS, 0, stream>>>(
        hid_bf, wqa_bf, b_qa, qc_f32, nullptr, nullptr, NROWS, Q_LR, HID_DIM);
    gemm_wmma<0><<<dim3((KV_LR + QK_ROPE + BN - 1) / BN, NROWS / BM), GEMM_THREADS, 0, stream>>>(
        hid_bf, wkva_bf, b_kva, kvc_f32, nullptr, nullptr, NROWS, KV_LR + QK_ROPE, HID_DIM);

    // 3) layernorms -> bf16 ; k RoPE broadcast straight to output
    ln_kernel<Q_LR, Q_LR><<<NROWS, 256, 0, stream>>>(qc_f32, g_qa_ln, b_qa_ln, qcln_bf);
    ln_kernel<KV_LR, KV_LR + QK_ROPE><<<NROWS, 256, 0, stream>>>(kvc_f32, g_kva_ln, b_kva_ln, kvlr_bf);
    krot_kernel<<<NROWS, 64, 0, stream>>>(kvc_f32, out);

    // 4) up-projections with fused head scatter
    gemm_wmma<1><<<dim3(NHEAD * QK_HEAD / BN, NROWS / BM), GEMM_THREADS, 0, stream>>>(
        qcln_bf, wqb_bf, b_qb, nullptr, out, qrope_f, NROWS, NHEAD * QK_HEAD, Q_LR);
    gemm_wmma<2><<<dim3(NHEAD * (QK_NOPE + V_DIM) / BN, NROWS / BM), GEMM_THREADS, 0, stream>>>(
        kvlr_bf, wkvb_bf, b_kvb, nullptr, out, nullptr, NROWS, NHEAD * (QK_NOPE + V_DIM), KV_LR);

    // 5) q RoPE epilogue
    qrope_kernel<<<NROWS, 256, 0, stream>>>(qrope_f, out);
}
